// EvasionGNN_24292335026414
// MI455X (gfx1250) — compile-verified, matrix operand order checked by reading, not run on hardware
//
#include <hip/hip_runtime.h>
#include <math.h>

#define HID   128
#define HEADS 4
#define CH    32
#define LAYERS 3
#define EPSV  1e-5f

// ---------------------------------------------------------------------------
// WMMA types (CDNA5 wave32): 16x16x32 bf16 -> f32
// ---------------------------------------------------------------------------
typedef __attribute__((ext_vector_type(16))) __bf16 bf16x16;
typedef __attribute__((ext_vector_type(8)))  float  fx8;

// Descriptor vector types for the TDM builtin (6-arg clang-23 form)
typedef __attribute__((ext_vector_type(4))) unsigned uintx4;
typedef __attribute__((ext_vector_type(4))) int      intx4;
typedef __attribute__((ext_vector_type(8))) int      intx8;

// Address-space-typed pointers for the CDNA5 async global->LDS builtins
typedef __attribute__((address_space(1))) int as1_int;
typedef __attribute__((address_space(3))) int as3_int;

// K-offset of VGPR slot v for the 16-bit A/B 16x32 layout (ISA 7.12.2):
// lanes 0-15: V0-3 -> K=0..7 (pairs), V4-7 -> K=16..23
// lanes16-31: V0-3 -> K=8..15,        V4-7 -> K=24..31
__device__ __forceinline__ int kb16(int v, int lane_hi8) {
  return ((v >> 2) << 4) + lane_hi8 + ((v & 3) << 1);
}

// Ordered-uint encoding of float for atomicMax
__device__ __forceinline__ unsigned fordU(float f) {
  unsigned u = __float_as_uint(f);
  return (u & 0x80000000u) ? ~u : (u | 0x80000000u);
}
__device__ __forceinline__ float unfordU(unsigned u) {
  return (u & 0x80000000u) ? __uint_as_float(u & 0x7fffffffu) : __uint_as_float(~u);
}

__device__ __forceinline__ float gelu_exact(float x) {
  return 0.5f * x * (1.0f + erff(x * 0.70710678118654752f));
}

// ---------------------------------------------------------------------------
// GEMM: out[M,Nout] = A[M,K] @ B[K,Nout] + bias[Nout]
// fp32 in, bf16 WMMA compute, f32 accumulate.
// Block = 256 threads (8 waves). Each block owns a 64-wide N panel; the raw
// fp32 K x 64 weight panel is DMA'd into LDS by the Tensor Data Mover (one
// descriptor, TENSORcnt-tracked) and converted to bf16 at fragment-read time.
// Each wave computes a 16(M) x 64(N) tile: 4 accumulators, A reused 4x.
// Requires M%16==0, Nout%64==0, K%32==0, K<=256 (all call sites comply).
// ---------------------------------------------------------------------------
__global__ __launch_bounds__(256)
void gemm_bf16_wmma_kernel(const float* __restrict__ A,
                           const float* __restrict__ Bm,
                           const float* __restrict__ bias,
                           float* __restrict__ out,
                           int M, int K, int Nout) {
  __shared__ float sRaw[256 * 64];          // fp32 K x 64 panel (K<=256), 64KB
  const int lane  = threadIdx.x & 31;
  const int wave  = threadIdx.x >> 5;
  const int ncol0 = blockIdx.y << 6;

  // ---- stage B panel into LDS ----
#if defined(__AMDGCN__) && __has_builtin(__builtin_amdgcn_tensor_load_to_lds) && __has_builtin(__builtin_amdgcn_s_wait_tensorcnt)
  if (threadIdx.x == 0) {                   // one wave issues the TDM descriptor
    const unsigned long long ga = (unsigned long long)(uintptr_t)(Bm + ncol0);
    const unsigned ldsOff = (unsigned)(uintptr_t)(void*)sRaw;  // LDS aperture: addr[31:0]
    uintx4 g0;
    g0[0] = 1u;                                         // count=1, user mode, no gather
    g0[1] = ldsOff;                                     // lds_addr (bytes)
    g0[2] = (unsigned)(ga & 0xffffffffu);               // global_addr[31:0] (tile start)
    g0[3] = (unsigned)((ga >> 32) & 0x01ffffffu)        // global_addr[56:32]
          | (2u << 30);                                 // type = 2 ("image")
    intx8 g1;
    g1[0] = (int)(2u << 16);                            // data_size=4B; mask=0
    g1[1] = (int)(((unsigned)Nout & 0xffffu) << 16);    // tensor_dim0[15:0]
    g1[2] = (int)(((unsigned)Nout >> 16)                // tensor_dim0[31:16]
          | (((unsigned)K & 0xffffu) << 16));           // tensor_dim1[15:0]
    g1[3] = (int)((((unsigned)K >> 16) & 0xffffu)       // tensor_dim1[31:16]
          | (64u << 16));                               // tile_dim0 = 64
    g1[4] = (int)((unsigned)K & 0xffffu);               // tile_dim1 = K; tile_dim2=0
    g1[5] = (int)Nout;                                  // tensor_dim0_stride[31:0]
    g1[6] = 0;                                          // stride hi / dim1_stride lo
    g1[7] = 0;                                          // dim1_stride hi (2D tensor)
    intx4 z4 = {0, 0, 0, 0};                            // groups 2/3 unused (2D)
    intx8 z8 = {0, 0, 0, 0, 0, 0, 0, 0};
    __builtin_amdgcn_tensor_load_to_lds(g0, g1, z4, z4, z8, 0);
  }
  __builtin_amdgcn_s_wait_tensorcnt(0);     // issuing wave waits; others pass
  __syncthreads();
#else
  for (int i = threadIdx.x; i < K * 64; i += blockDim.x)
    sRaw[i] = Bm[(size_t)(i >> 6) * Nout + ncol0 + (i & 63)];
  __syncthreads();
#endif

  const int mt = blockIdx.x * (blockDim.x >> 5) + wave;
  if (mt >= (M >> 4)) return;
  const int lh  = (lane >= 16) ? 8 : 0;
  const int l15 = lane & 15;
  const int row = (mt << 4) + l15;

  fx8 acc[4] = {};
  for (int k0 = 0; k0 < K; k0 += 32) {
    bf16x16 a;
    bf16x16 b[4];
    const float* arow = A + (size_t)row * K + k0;
#pragma unroll
    for (int v = 0; v < 8; ++v) {
      const int k = kb16(v, lh);
      float2 p = *(const float2*)(arow + k);        // contiguous K pair
      a[2 * v]     = (__bf16)p.x;
      a[2 * v + 1] = (__bf16)p.y;
      const int krow = (k0 + k) << 6;               // row offset in sRaw
#pragma unroll
      for (int j = 0; j < 4; ++j) {
        const int c = (j << 4) + l15;               // ds_load_2addr pair (k, k+1)
        b[j][2 * v]     = (__bf16)sRaw[krow + c];
        b[j][2 * v + 1] = (__bf16)sRaw[krow + 64 + c];
      }
    }
#pragma unroll
    for (int j = 0; j < 4; ++j)                     // 4x v_wmma per K-step
      acc[j] = __builtin_amdgcn_wmma_f32_16x16x32_bf16(
          false, a, false, b[j], (short)0, acc[j], false, false);
  }
#pragma unroll
  for (int j = 0; j < 4; ++j) {
    const int col = ncol0 + (j << 4) + l15;
    const float bv = bias ? bias[col] : 0.0f;
#pragma unroll
    for (int r = 0; r < 8; ++r)                     // C/D: VGPR r -> M=r(+8 hi)
      out[(size_t)((mt << 4) + lh + r) * Nout + col] = acc[j][r] + bv;
  }
}

// ---------------------------------------------------------------------------
// Fills
// ---------------------------------------------------------------------------
__global__ void fill_f32_kernel(float* p, float v, int n) {
  int i = blockIdx.x * blockDim.x + threadIdx.x;
  if (i < n) p[i] = v;
}
__global__ void fill_u32_kernel(unsigned* p, unsigned v, int n) {
  int i = blockIdx.x * blockDim.x + threadIdx.x;
  if (i < n) p[i] = v;
}

// ---------------------------------------------------------------------------
// Per-node attention scores: es/ed[n,h] = sum_c hp[n,h,c] * att[h,c]
// ---------------------------------------------------------------------------
__global__ void att_scores_kernel(const float* __restrict__ hp,
                                  const float* __restrict__ asrc,
                                  const float* __restrict__ adst,
                                  float* __restrict__ es,
                                  float* __restrict__ ed, int n) {
  int t = blockIdx.x * blockDim.x + threadIdx.x;
  if (t >= n * HEADS) return;
  const int nd = t >> 2, hh = t & 3;
  const float* hv = hp + (size_t)nd * HID + hh * CH;
  float s = 0.f, d = 0.f;
#pragma unroll
  for (int c = 0; c < CH; ++c) {
    float x = hv[c];
    s += x * asrc[hh * CH + c];
    d += x * adst[hh * CH + c];
  }
  es[t] = s;
  ed[t] = d;
}

// ---------------------------------------------------------------------------
// Stage a block's 64 edge (src,dst) index pairs into LDS. Four threads share
// each edge, so this removes 4x redundant global gathers. Uses the CDNA5
// async global->LDS path (ASYNCcnt-tracked DMA) when available.
// ---------------------------------------------------------------------------
__device__ __forceinline__ void stage_edges(const int* __restrict__ ei,
                                            int E, int Et, int ebase,
                                            int* ssrc, int* sdst) {
  const int t = threadIdx.x;
  if (t == 0 && ebase + 4096 < E) __builtin_prefetch(ei + ebase + 4096, 0, 1);
  if (t < 64) {
    const int e = ebase + t;
    if (e < E) {
#if defined(__AMDGCN__) && __has_builtin(__builtin_amdgcn_global_load_async_to_lds_b32) && __has_builtin(__builtin_amdgcn_s_wait_asynccnt)
      __builtin_amdgcn_global_load_async_to_lds_b32(
          (as1_int*)(uintptr_t)(ei + e),
          (as3_int*)(uintptr_t)(ssrc + t), 0, 0);
      __builtin_amdgcn_global_load_async_to_lds_b32(
          (as1_int*)(uintptr_t)(ei + E + e),
          (as3_int*)(uintptr_t)(sdst + t), 0, 0);
#else
      ssrc[t] = ei[e];
      sdst[t] = ei[E + e];
#endif
    } else if (e < Et) {
      ssrc[t] = e - E;            // self loop
      sdst[t] = e - E;
    }
  }
#if defined(__AMDGCN__) && __has_builtin(__builtin_amdgcn_global_load_async_to_lds_b32) && __has_builtin(__builtin_amdgcn_s_wait_asynccnt)
  __builtin_amdgcn_s_wait_asynccnt(0);
#endif
  __syncthreads();
}

// ---------------------------------------------------------------------------
// Edge pass 1: segment max of leaky_relu(es[src]+ed[dst]) over dst
// 256 threads = 64 edges x 4 heads per block
// ---------------------------------------------------------------------------
__global__ __launch_bounds__(256)
void edge_max_kernel(const int* __restrict__ ei, int E, int n,
                     const float* __restrict__ es,
                     const float* __restrict__ ed,
                     unsigned* __restrict__ mu) {
  __shared__ int ssrc[64], sdst[64];
  const int Et = E + n;
  const int ebase = blockIdx.x * 64;
  stage_edges(ei, E, Et, ebase, ssrc, sdst);
  const int le = threadIdx.x >> 2, hh = threadIdx.x & 3;
  const int e = ebase + le;
  if (e >= Et) return;
  const int s = ssrc[le], d = sdst[le];
  float v = es[s * HEADS + hh] + ed[d * HEADS + hh];
  v = v > 0.f ? v : 0.2f * v;
  atomicMax(&mu[d * HEADS + hh], fordU(v));
}

// ---------------------------------------------------------------------------
// Edge pass 2: ee = exp(e - m[dst]); denom[dst] += ee; stash ee
// ---------------------------------------------------------------------------
__global__ __launch_bounds__(256)
void edge_soft_kernel(const int* __restrict__ ei, int E, int n,
                      const float* __restrict__ es,
                      const float* __restrict__ ed,
                      const unsigned* __restrict__ mu,
                      float* __restrict__ eebuf,
                      float* __restrict__ denom) {
  __shared__ int ssrc[64], sdst[64];
  const int Et = E + n;
  const int ebase = blockIdx.x * 64;
  stage_edges(ei, E, Et, ebase, ssrc, sdst);
  const int le = threadIdx.x >> 2, hh = threadIdx.x & 3;
  const int e = ebase + le;
  if (e >= Et) return;
  const int s = ssrc[le], d = sdst[le];
  float v = es[s * HEADS + hh] + ed[d * HEADS + hh];
  v = v > 0.f ? v : 0.2f * v;
  const float m = unfordU(mu[d * HEADS + hh]);
  const float ee = __expf(v - m);
  eebuf[e * HEADS + hh] = ee;
  atomicAdd(&denom[d * HEADS + hh], ee);
}

// ---------------------------------------------------------------------------
// Edge pass 3: hnew[dst,:] += alpha_h * hp[src,:]
// One wave per edge: 32 lanes x 4 features each, fully coalesced atomics.
// ---------------------------------------------------------------------------
__global__ __launch_bounds__(256)
void edge_agg_kernel(const int* __restrict__ ei, int E, int n,
                     const float* __restrict__ eebuf,
                     const float* __restrict__ denom,
                     const float* __restrict__ hp,
                     float* __restrict__ hnew) {
  const int Et = E + n;
  const int lane = threadIdx.x & 31;
  const int e = blockIdx.x * (blockDim.x >> 5) + (threadIdx.x >> 5);
  if (e >= Et) return;
  int s, d;
  if (e < E) { s = ei[e]; d = ei[E + e]; } else { s = d = e - E; }
  float alpha[HEADS];
#pragma unroll
  for (int j = 0; j < HEADS; ++j)
    alpha[j] = eebuf[e * HEADS + j] / (denom[d * HEADS + j] + 1e-16f);
  const float* hs = hp   + (size_t)s * HID;
  float*       hd = hnew + (size_t)d * HID;
#pragma unroll
  for (int j = 0; j < HEADS; ++j) {
    const int f = (j << 5) + lane;                  // feature; head = j
    atomicAdd(&hd[f], alpha[j] * hs[f]);
  }
}

// ---------------------------------------------------------------------------
// BatchNorm stats over N for x = h + hnew + gat_bias (wave32 shuffle reduce)
// ---------------------------------------------------------------------------
__global__ void bn_stats_kernel(const float* __restrict__ h,
                                const float* __restrict__ hnew,
                                const float* __restrict__ gbias,
                                float* __restrict__ sum,
                                float* __restrict__ sq, int n) {
  const int f = blockIdx.y;
  const int nd = blockIdx.x * blockDim.x + threadIdx.x;
  float x = 0.f, x2 = 0.f;
  if (nd < n) {
    float v = h[(size_t)nd * HID + f] + hnew[(size_t)nd * HID + f] + gbias[f];
    x = v; x2 = v * v;
  }
#pragma unroll
  for (int o = 16; o > 0; o >>= 1) {                // wave32 reduction
    x  += __shfl_xor(x,  o, 32);
    x2 += __shfl_xor(x2, o, 32);
  }
  if ((threadIdx.x & 31) == 0) {
    atomicAdd(&sum[f], x);
    atomicAdd(&sq[f],  x2);
  }
}

__global__ void bn_finalize_kernel(const float* sum, const float* sq,
                                   float* mu, float* rs, int n) {
  const int f = threadIdx.x;
  float m = sum[f] / (float)n;
  float v = sq[f] / (float)n - m * m;
  mu[f] = m;
  rs[f] = rsqrtf(v + EPSV);
}

__global__ void bn_apply_kernel(float* __restrict__ h,
                                const float* __restrict__ hnew,
                                const float* __restrict__ gbias,
                                const float* __restrict__ mu,
                                const float* __restrict__ rs,
                                const float* __restrict__ g,
                                const float* __restrict__ b, int n) {
  int t = blockIdx.x * blockDim.x + threadIdx.x;
  if (t >= n * HID) return;
  const int f = t & (HID - 1);
  float x = h[t] + hnew[t] + gbias[f];
  float y = (x - mu[f]) * rs[f] * g[f] + b[f];
  h[t] = y > 0.f ? y : 0.f;
}

// ---------------------------------------------------------------------------
// Graph pooling (mean + max) over sorted batch ids
// ---------------------------------------------------------------------------
__global__ void pool_kernel(const float* __restrict__ h,
                            const int* __restrict__ batch,
                            float* __restrict__ meanp,
                            unsigned* __restrict__ maxu,
                            float* __restrict__ cnt, int n) {
  int t = blockIdx.x * blockDim.x + threadIdx.x;
  if (t >= n * HID) return;
  const int nd = t >> 7, f = t & (HID - 1);
  const int bg = batch[nd];
  float v = h[t];
  atomicAdd(&meanp[bg * HID + f], v);
  atomicMax(&maxu[bg * HID + f], fordU(v));
  if (f == 0) atomicAdd(&cnt[bg], 1.0f);
}

__global__ void gf_kernel(const float* meanp, const unsigned* maxu,
                          const float* cnt, float* gf, int Bq) {
  int t = blockIdx.x * blockDim.x + threadIdx.x;
  if (t >= Bq * 2 * HID) return;
  const int b = t / (2 * HID), f = t % (2 * HID);
  gf[t] = (f < HID) ? meanp[b * HID + f] / cnt[b]
                    : unfordU(maxu[b * HID + (f - HID)]);
}

// ---------------------------------------------------------------------------
// LayerNorm + exact GELU (one block per row, blockDim == D, D<=256, pow2)
// ---------------------------------------------------------------------------
__global__ void ln_gelu_kernel(float* __restrict__ x,
                               const float* __restrict__ g,
                               const float* __restrict__ b, int D) {
  __shared__ float sm[256];
  const int r = blockIdx.x, f = threadIdx.x;
  float v = x[(size_t)r * D + f];
  sm[f] = v; __syncthreads();
  for (int o = D >> 1; o > 0; o >>= 1) { if (f < o) sm[f] += sm[f + o]; __syncthreads(); }
  const float muv = sm[0] / (float)D; __syncthreads();
  const float dv = v - muv;
  sm[f] = dv * dv; __syncthreads();
  for (int o = D >> 1; o > 0; o >>= 1) { if (f < o) sm[f] += sm[f + o]; __syncthreads(); }
  const float var = sm[0] / (float)D;
  float y = dv * rsqrtf(var + EPSV) * g[f] + b[f];
  x[(size_t)r * D + f] = gelu_exact(y);
}

__global__ void gelu_kernel(float* x, int n) {
  int i = blockIdx.x * blockDim.x + threadIdx.x;
  if (i < n) x[i] = gelu_exact(x[i]);
}

// Small dense head: out[Bq,O] = x[Bq,K] @ w[K,O] + b  (O==4, VALU is fine)
__global__ void small_dense_kernel(const float* __restrict__ x,
                                   const float* __restrict__ w,
                                   const float* __restrict__ b,
                                   float* __restrict__ out,
                                   int Bq, int K, int O) {
  int t = blockIdx.x * blockDim.x + threadIdx.x;
  if (t >= Bq * O) return;
  const int r = t / O, o = t % O;
  float acc = b[o];
  for (int k = 0; k < K; ++k) acc += x[(size_t)r * K + k] * w[k * O + o];
  out[t] = acc;
}

// ---------------------------------------------------------------------------
// Host-side orchestration
// ---------------------------------------------------------------------------
static inline int cdiv(int a, int b) { return (a + b - 1) / b; }

extern "C" void kernel_launch(void* const* d_in, const int* in_sizes, int n_in,
                              void* d_out, int out_size, void* d_ws, size_t ws_size,
                              hipStream_t stream) {
  const float* x        = (const float*)d_in[0];
  const int*   eidx     = (const int*)  d_in[1];
  const int*   batch    = (const int*)  d_in[2];
  const float* proj_w   = (const float*)d_in[3];
  const float* proj_b   = (const float*)d_in[4];
  const float* lin_w    = (const float*)d_in[5];
  const float* att_src  = (const float*)d_in[6];
  const float* att_dst  = (const float*)d_in[7];
  const float* gat_bias = (const float*)d_in[8];
  const float* bn_g     = (const float*)d_in[9];
  const float* bn_b     = (const float*)d_in[10];
  const float* cls_w1   = (const float*)d_in[11];
  const float* cls_b1   = (const float*)d_in[12];
  const float* ln1_g    = (const float*)d_in[13];
  const float* ln1_b    = (const float*)d_in[14];
  const float* cls_w2   = (const float*)d_in[15];
  const float* cls_b2   = (const float*)d_in[16];
  const float* ln2_g    = (const float*)d_in[17];
  const float* ln2_b    = (const float*)d_in[18];
  const float* cls_w3   = (const float*)d_in[19];
  const float* cls_b3   = (const float*)d_in[20];
  const float* sop_w1   = (const float*)d_in[21];
  const float* sop_b1   = (const float*)d_in[22];
  const float* sop_w2   = (const float*)d_in[23];
  const float* sop_b2   = (const float*)d_in[24];
  float* out = (float*)d_out;

  const int N  = in_sizes[0] / 32;      // 100000
  const int E  = in_sizes[1] / 2;       // 1600000
  const int Et = E + N;                 // + self loops
  const int Bq = out_size / 8;          // 64 graphs (two [B,4] heads)
  const int FIN = 32;

  // ---- workspace carve ----
  char* wp = (char*)d_ws;
  auto carve = [&](size_t bytes) -> void* {
    void* p = (void*)wp;
    wp += (bytes + 255) & ~(size_t)255;
    return p;
  };
  float*    h     = (float*)   carve((size_t)N * HID * 4);
  float*    hp    = (float*)   carve((size_t)N * HID * 4);
  float*    hnew  = (float*)   carve((size_t)N * HID * 4);
  float*    es    = (float*)   carve((size_t)N * HEADS * 4);
  float*    ed    = (float*)   carve((size_t)N * HEADS * 4);
  unsigned* m_u   = (unsigned*)carve((size_t)N * HEADS * 4);
  float*    denom = (float*)   carve((size_t)N * HEADS * 4);
  float*    eebuf = (float*)   carve((size_t)Et * HEADS * 4);
  float*    bnsum = (float*)   carve(HID * 4);
  float*    bnsq  = (float*)   carve(HID * 4);
  float*    bnmu  = (float*)   carve(HID * 4);
  float*    bnrs  = (float*)   carve(HID * 4);
  float*    cnt   = (float*)   carve((size_t)Bq * 4);
  float*    meanp = (float*)   carve((size_t)Bq * HID * 4);
  unsigned* maxu  = (unsigned*)carve((size_t)Bq * HID * 4);
  float*    gf    = (float*)   carve((size_t)Bq * 2 * HID * 4);
  float*    t1    = (float*)   carve((size_t)Bq * 256 * 4);
  float*    t2    = (float*)   carve((size_t)Bq * 128 * 4);
  float*    s1    = (float*)   carve((size_t)Bq * 64 * 4);
  (void)ws_size; (void)n_in;

  const unsigned NEG_INF_ORD = 0x007FFFFFu;  // fordU(-inf)

  auto gemm = [&](const float* A, const float* Bm, const float* bias,
                  float* O, int M, int K, int Nout) {
    dim3 grid(cdiv(M / 16, 8), Nout / 64);
    gemm_bf16_wmma_kernel<<<grid, 256, 0, stream>>>(A, Bm, bias, O, M, K, Nout);
  };

  // ---- h = x @ proj_w + proj_b ----
  gemm(x, proj_w, proj_b, h, N, FIN, HID);

  // ---- 3 GAT layers ----
  for (int L = 0; L < LAYERS; ++L) {
    const float* lw = lin_w    + (size_t)L * HID * HID;
    const float* as = att_src  + (size_t)L * HEADS * CH;
    const float* ad = att_dst  + (size_t)L * HEADS * CH;
    const float* gb = gat_bias + (size_t)L * HID;

    fill_f32_kernel<<<cdiv(N * HID, 256), 256, 0, stream>>>(hnew, 0.f, N * HID);
    fill_u32_kernel<<<cdiv(N * HEADS, 256), 256, 0, stream>>>(m_u, NEG_INF_ORD, N * HEADS);
    fill_f32_kernel<<<cdiv(N * HEADS, 256), 256, 0, stream>>>(denom, 0.f, N * HEADS);
    fill_f32_kernel<<<1, HID, 0, stream>>>(bnsum, 0.f, HID);
    fill_f32_kernel<<<1, HID, 0, stream>>>(bnsq, 0.f, HID);

    gemm(h, lw, nullptr, hp, N, HID, HID);                         // hp = h @ W
    att_scores_kernel<<<cdiv(N * HEADS, 256), 256, 0, stream>>>(hp, as, ad, es, ed, N);

    edge_max_kernel <<<cdiv(Et, 64), 256, 0, stream>>>(eidx, E, N, es, ed, m_u);
    edge_soft_kernel<<<cdiv(Et, 64), 256, 0, stream>>>(eidx, E, N, es, ed, m_u, eebuf, denom);
    edge_agg_kernel <<<cdiv(Et, 8), 256, 0, stream>>>(eidx, E, N, eebuf, denom, hp, hnew);

    dim3 sgrid(cdiv(N, 256), HID);
    bn_stats_kernel<<<sgrid, 256, 0, stream>>>(h, hnew, gb, bnsum, bnsq, N);
    bn_finalize_kernel<<<1, HID, 0, stream>>>(bnsum, bnsq, bnmu, bnrs, N);
    bn_apply_kernel<<<cdiv(N * HID, 256), 256, 0, stream>>>(
        h, hnew, gb, bnmu, bnrs, bn_g + (size_t)L * HID, bn_b + (size_t)L * HID, N);
  }

  // ---- pooling ----
  fill_f32_kernel<<<1, Bq, 0, stream>>>(cnt, 0.f, Bq);
  fill_f32_kernel<<<cdiv(Bq * HID, 256), 256, 0, stream>>>(meanp, 0.f, Bq * HID);
  fill_u32_kernel<<<cdiv(Bq * HID, 256), 256, 0, stream>>>(maxu, NEG_INF_ORD, Bq * HID);
  pool_kernel<<<cdiv(N * HID, 256), 256, 0, stream>>>(h, batch, meanp, maxu, cnt, N);
  gf_kernel<<<cdiv(Bq * 2 * HID, 256), 256, 0, stream>>>(meanp, maxu, cnt, gf, Bq);

  // ---- evasion head ----
  gemm(gf, cls_w1, cls_b1, t1, Bq, 256, 256);
  ln_gelu_kernel<<<Bq, 256, 0, stream>>>(t1, ln1_g, ln1_b, 256);
  gemm(t1, cls_w2, cls_b2, t2, Bq, 256, 128);
  ln_gelu_kernel<<<Bq, 128, 0, stream>>>(t2, ln2_g, ln2_b, 128);
  small_dense_kernel<<<cdiv(Bq * 4, 64), 64, 0, stream>>>(t2, cls_w3, cls_b3, out, Bq, 128, 4);

  // ---- sophistication head ----
  gemm(gf, sop_w1, sop_b1, s1, Bq, 256, 64);
  gelu_kernel<<<cdiv(Bq * 64, 256), 256, 0, stream>>>(s1, Bq * 64);
  small_dense_kernel<<<cdiv(Bq * 4, 64), 64, 0, stream>>>(s1, sop_w2, sop_b2, out + Bq * 4, Bq, 64, 4);
}
